// Seq2Seq_44032004719086
// MI455X (gfx1250) — matrix operand order repaired and checked by reading, not compiled, not run on hardware
//
#include <hip/hip_runtime.h>
#include <hip/hip_bf16.h>

typedef __attribute__((ext_vector_type(16))) _Float16 v16h;
typedef __attribute__((ext_vector_type(8)))  float    v8f;
typedef const __attribute__((address_space(1))) v16h* gvp;   // force global_load_b128

#define NWG 16
#define NTH 256
#define HDIM 512
#define TSTEPS 28

__device__ __forceinline__ float sigf(float x) { return 1.f / (1.f + __expf(-x)); }

// ---------------------------------------------------------------------------
// Prep: swizzle [Wih | Whh] (f32) into f16 WMMA A-fragment order.
// Fragment (g = 16-row group, c = 32-K chunk) = 512 halves:
//   half index = lane*16 + j, M = lane&15,
//   K = c*32 + (lane<16 ? 0 : 16) + j
// ---------------------------------------------------------------------------
__global__ void prep_frag(_Float16* __restrict__ dst,
                          const float* __restrict__ Wih,
                          const float* __restrict__ Whh,
                          int in_dim, int C) {
  int idx = blockIdx.x * NTH + threadIdx.x;
  int total = 128 * C * 512;
  if (idx >= total) return;
  int within = idx & 511;
  int fragIdx = idx >> 9;
  int lane = within >> 4, j = within & 15;
  int g = fragIdx / C, c = fragIdx % C;
  int row = g * 16 + (lane & 15);
  int K = c * 32 + ((lane < 16) ? 0 : 16) + j;
  float v = 0.f;
  if (K < in_dim)            v = Wih[(size_t)row * in_dim + K];
  else if (K < in_dim + 512) v = Whh[(size_t)row * 512 + (K - in_dim)];
  dst[idx] = (_Float16)v;
}

__global__ void prep_misc(float* __restrict__ benc, float* __restrict__ bdec,
                          const float* ebih, const float* ebhh,
                          const float* dbih, const float* dbhh,
                          _Float16* __restrict__ hb, unsigned* __restrict__ cnt) {
  int idx = blockIdx.x * NTH + threadIdx.x;
  if (idx < 3 * 2048) {
    benc[idx] = ebih[idx] + ebhh[idx];
    bdec[idx] = dbih[idx] + dbhh[idx];
  }
  if (idx < 2 * 3 * HDIM) hb[idx] = (_Float16)0.f;
  if (idx == 0) *cnt = 0u;
}

// ---------------------------------------------------------------------------
// Persistent kernel: whole encoder + 28 decoder steps. 16 WGs, L2-atomic
// grid barrier (all WGs trivially resident).
// ---------------------------------------------------------------------------
__device__ __forceinline__ void grid_sync(unsigned* cnt, int& epoch) {
  __syncthreads();
  if (threadIdx.x == 0) {
    ++epoch;
    __threadfence();
    __hip_atomic_fetch_add(cnt, 1u, __ATOMIC_RELEASE, __HIP_MEMORY_SCOPE_AGENT);
    while (__hip_atomic_load(cnt, __ATOMIC_ACQUIRE, __HIP_MEMORY_SCOPE_AGENT) <
           (unsigned)epoch * NWG)
      __builtin_amdgcn_s_sleep(1);
  }
  __syncthreads();
}

// GEMV core: 16 gate rows per wave, K = C*32. W rows live in the A operand
// (prep-swizzled fragments, global_load_b128), activation vector replicated
// across B columns (ds_load_b128 from LDS). Full f32 accumulate in C regs.
template <int C>
__device__ __forceinline__ v8f gemv(const _Float16* Wm, const _Float16* xh,
                                    int lane) {
  const _Float16* bl = xh + ((lane >> 4) << 4);
  v8f acc = {};
#pragma unroll
  for (int c = 0; c < C; ++c) {
    v16h a = *(gvp)(unsigned long long)(Wm + (size_t)c * 512 + lane * 16);
    v16h b = *(const v16h*)(bl + c * 32);
    acc = __builtin_amdgcn_wmma_f32_16x16x32_f16(
        false, a, false, b, (short)0, acc, false, false);
  }
  return acc;
}

struct SharedBlk {
  _Float16 xh[1088];     // padded concat input, f16, 32B-aligned chunks
  float gates[4][32];    // this WG's 32 rows per gate
  float cst[3][32];      // cell state (WG-private, lives whole kernel)
  float pr[32][8];       // output-GEMV partials
  float y[32];           // current y (feedback), first 21 valid
};

__global__ __launch_bounds__(NTH) void seq2seq_main(
    const _Float16* __restrict__ We0, const _Float16* __restrict__ We1,
    const _Float16* __restrict__ We2, const _Float16* __restrict__ Wd0,
    const _Float16* __restrict__ Wd1, const _Float16* __restrict__ Wd2,
    const float* __restrict__ benc, const float* __restrict__ bdec,
    _Float16* __restrict__ hb,        // [2][3][512] double-buffered h (f16)
    _Float16* __restrict__ encout,    // [512]
    unsigned* __restrict__ cnt,
    const float* __restrict__ x_in, const float* __restrict__ prevy,
    const float* __restrict__ outW, const float* __restrict__ outb,
    float* __restrict__ out) {
  __shared__ __align__(32) SharedBlk s;
  const int tid = threadIdx.x, w = blockIdx.x;
  const int lane = tid & 31, wv = tid >> 5;
  const int q = wv >> 1, sub = wv & 1;          // gate, half-of-32-rows
  const int g = q * 32 + w * 2 + sub;           // global 16-row group
  int epoch = 0;

  if (tid < 32) { s.cst[0][tid] = 0.f; s.cst[1][tid] = 0.f; s.cst[2][tid] = 0.f; }
  if (tid < 21) s.y[tid] = prevy[tid];
  __syncthreads();

  const int Cn[6] = {17, 32, 32, 33, 32, 32};   // K-chunks (K padded to 32)

  for (int t = 0; t <= TSTEPS; ++t) {
    const int p = t & 1, pw = 1 - p;            // read side / write side
    const bool enc = (t == 0);
    for (int l = 0; l < 3; ++l) {
      const int mi = (enc ? 0 : 3) + l;
      const int C = Cn[mi];
      const int Kp = C * 32;
      // --- assemble concat input [x | h_l(prev)] as f16 in LDS ---
      for (int k = tid; k < Kp; k += NTH) {
        float v;
        if (l == 0) {
          if (enc)
            v = (k < 21) ? x_in[k]
              : ((k < 533) ? (float)hb[(p * 3 + 0) * HDIM + (k - 21)] : 0.f);
          else
            v = (k < 21) ? s.y[k]
              : ((k < 533) ? (float)encout[k - 21]
              : ((k < 1045) ? (float)hb[(p * 3 + 0) * HDIM + (k - 533)] : 0.f));
        } else {
          v = (k < HDIM) ? (float)hb[(pw * 3 + (l - 1)) * HDIM + k]
                         : (float)hb[(p * 3 + l) * HDIM + (k - 512)];
        }
        s.xh[k] = (_Float16)v;
      }
      __syncthreads();
      // --- GEMV via WMMA (per-matrix dispatch keeps loads global + unrolled)
      v8f acc;
      switch (mi) {
        case 0:  acc = gemv<17>(We0 + (size_t)g * 17 * 512, s.xh, lane); break;
        case 1:  acc = gemv<32>(We1 + (size_t)g * 32 * 512, s.xh, lane); break;
        case 2:  acc = gemv<32>(We2 + (size_t)g * 32 * 512, s.xh, lane); break;
        case 3:  acc = gemv<33>(Wd0 + (size_t)g * 33 * 512, s.xh, lane); break;
        case 4:  acc = gemv<32>(Wd1 + (size_t)g * 32 * 512, s.xh, lane); break;
        default: acc = gemv<32>(Wd2 + (size_t)g * 32 * 512, s.xh, lane); break;
      }
      // C/D layout: VGPR r holds M=r (lanes 0-15) / M=8+r (lanes 16-31),
      // replicated across N; extract at N==0 (lanes 0 and 16).
      if ((lane & 15) == 0) {
        const int base = sub * 16 + ((lane >> 4) << 3);
#pragma unroll
        for (int r = 0; r < 8; ++r) s.gates[q][base + r] = acc[r];
      }
      __syncthreads();
      // --- pointwise LSTM update for this WG's 32 h-dims ---
      if (tid < 32) {
        const int r512 = w * 32 + tid;
        const float* bl2 = (enc ? benc : bdec) + l * 2048;
        float gi = s.gates[0][tid] + bl2[0 * HDIM + r512];
        float gf = s.gates[1][tid] + bl2[1 * HDIM + r512];
        float gg = s.gates[2][tid] + bl2[2 * HDIM + r512];
        float go = s.gates[3][tid] + bl2[3 * HDIM + r512];
        float c2 = sigf(gf) * s.cst[l][tid] + sigf(gi) * tanhf(gg);
        float h2 = sigf(go) * tanhf(c2);
        s.cst[l][tid] = c2;
        hb[(pw * 3 + l) * HDIM + r512] = (_Float16)h2;
        if (enc && l == 2) encout[r512] = (_Float16)h2;
      }
      grid_sync(cnt, epoch);
    }
    if (!enc) {
      // y = [h3 | enc_out] @ out_W.T + out_b   (redundant per WG, no barrier)
      const int r = tid & 31, sl = tid >> 5;
      float accp = 0.f;
      if (r < 21) {
        const float* wr = outW + r * 1024 + sl * 128;
        for (int k = 0; k < 128; ++k) {
          int kk = sl * 128 + k;
          float xv = (kk < HDIM) ? (float)hb[(pw * 3 + 2) * HDIM + kk]
                                 : (float)encout[kk - HDIM];
          accp += xv * wr[k];
        }
      }
      s.pr[r][sl] = accp;
      __syncthreads();
      if (tid < 21) {
        float a2 = outb[tid];
#pragma unroll
        for (int sx = 0; sx < 8; ++sx) a2 += s.pr[tid][sx];
        s.y[tid] = a2;
        if (w == 0) out[(t - 1) * 21 + tid] = a2;
      }
      __syncthreads();
    }
  }
}

extern "C" void kernel_launch(void* const* d_in, const int* in_sizes, int n_in,
                              void* d_out, int out_size, void* d_ws, size_t ws_size,
                              hipStream_t stream) {
  const float* x      = (const float*)d_in[0];
  const float* prev_y = (const float*)d_in[1];
  const float* eW0    = (const float*)d_in[2];
  const float* eW1    = (const float*)d_in[3];
  const float* eW2    = (const float*)d_in[4];
  const float* eWhh   = (const float*)d_in[5];
  const float* ebih   = (const float*)d_in[6];
  const float* ebhh   = (const float*)d_in[7];
  // d_in[8..10] = attn_W, attn_b, v_W: provably dead (softmax over len-1 == 1)
  const float* dW0    = (const float*)d_in[11];
  const float* dW1    = (const float*)d_in[12];
  const float* dW2    = (const float*)d_in[13];
  const float* dWhh   = (const float*)d_in[14];
  const float* dbih   = (const float*)d_in[15];
  const float* dbhh   = (const float*)d_in[16];
  const float* outW   = (const float*)d_in[17];
  const float* outb   = (const float*)d_in[18];
  float* out = (float*)d_out;

  // ---- workspace layout ----
  size_t off = 0;
  auto take = [&](size_t bytes) { size_t o = off; off = (off + bytes + 255) & ~(size_t)255; return o; };
  const int Cs[6] = {17, 32, 32, 33, 32, 32};
  size_t woff[6];
  for (int m = 0; m < 6; ++m) woff[m] = take((size_t)128 * Cs[m] * 512 * sizeof(_Float16));
  size_t benc_o = take(3 * 2048 * sizeof(float));
  size_t bdec_o = take(3 * 2048 * sizeof(float));
  size_t hb_o   = take(2 * 3 * HDIM * sizeof(_Float16));
  size_t enc_o  = take(HDIM * sizeof(_Float16));
  size_t cnt_o  = take(256);
  if (off > ws_size) return;  // insufficient scratch

  char* ws = (char*)d_ws;
  _Float16* Wf[6];
  for (int m = 0; m < 6; ++m) Wf[m] = (_Float16*)(ws + woff[m]);
  float* benc = (float*)(ws + benc_o);
  float* bdec = (float*)(ws + bdec_o);
  _Float16* hb = (_Float16*)(ws + hb_o);
  _Float16* encout = (_Float16*)(ws + enc_o);
  unsigned* cnt = (unsigned*)(ws + cnt_o);

  // ---- weight prep (f32 -> f16, WMMA A-fragment swizzle) ----
  const float* WihP[6] = {eW0, eW1, eW2, dW0, dW1, dW2};
  const float* WhhP[6] = {eWhh, eWhh + 2048 * 512, eWhh + 2 * 2048 * 512,
                          dWhh, dWhh + 2048 * 512, dWhh + 2 * 2048 * 512};
  const int inD[6] = {21, 512, 512, 533, 512, 512};
  for (int m = 0; m < 6; ++m) {
    int total = 128 * Cs[m] * 512;
    hipLaunchKernelGGL(prep_frag, dim3((total + NTH - 1) / NTH), dim3(NTH), 0, stream,
                       Wf[m], WihP[m], WhhP[m], inD[m], Cs[m]);
  }
  hipLaunchKernelGGL(prep_misc, dim3((3 * 2048 + NTH - 1) / NTH), dim3(NTH), 0, stream,
                     benc, bdec, ebih, ebhh, dbih, dbhh, hb, cnt);

  // ---- persistent recurrence kernel ----
  hipLaunchKernelGGL(seq2seq_main, dim3(NWG), dim3(NTH), 0, stream,
                     Wf[0], Wf[1], Wf[2], Wf[3], Wf[4], Wf[5],
                     benc, bdec, hb, encout, cnt,
                     x, prev_y, outW, outb, out);
}
